// GraphAttentionDecoder_80650895884835
// MI455X (gfx1250) — matrix operand order repaired and checked by reading, not compiled, not run on hardware
//
#include <hip/hip_runtime.h>
#include <hip/hip_bf16.h>
#include <stdint.h>

// ---------------------------------------------------------------------------
// GATv2 encoder + GCN decoder for MI455X (gfx1250, wave32, WMMA).
// Dense GEMMs: v_wmma_f32_16x16x32_f16, A-fragment reused across 2-4 WMMAs,
// single-wave blocks with __launch_bounds__(32,1) to avoid scratch spills.
// Edge gather/scatter kernels: wave-per-edge, float4 (b128) loads, L2-resident.
// ---------------------------------------------------------------------------

typedef __attribute__((ext_vector_type(16))) _Float16 v16h;
typedef __attribute__((ext_vector_type(8)))  float    v8f;

#define NH     4
#define NC     64
#define DIN    32
#define DHID   256   // H*C
#define DOUT   128
#define NEG_SLOPE 0.2f

// ---- order-preserving float<->uint encoding for atomicMax-based segment max
__device__ __forceinline__ unsigned enc_f32(float f) {
  unsigned u = __float_as_uint(f);
  return (u & 0x80000000u) ? ~u : (u | 0x80000000u);
}
__device__ __forceinline__ float dec_f32(unsigned u) {
  return __uint_as_float((u & 0x80000000u) ? (u ^ 0x80000000u) : ~u);
}

// ---------------------------------------------------------------------------
// Zero-fill helper (graph-capture safe; no hipMemset*).
__global__ void zero_f32_kernel(float* __restrict__ p, long n) {
  long i = (long)blockIdx.x * blockDim.x + threadIdx.x;
  long stride = (long)gridDim.x * blockDim.x;
  for (; i < n; i += stride) p[i] = 0.0f;
}

// ---------------------------------------------------------------------------
// WMMA fragment helpers (contiguous-K A/B order chosen by this code, not the ISA table order; wave32):
//   A (16x32 f16): lanes 0-15 = rows M=0..15 holding K 0..15,
//                  lanes 16-31 = same rows holding K 16..31.
//   B (32x16 f16): lanes 0-15 = cols N=0..15 holding K 0..15,
//                  lanes 16-31 = same cols holding K 16..31.
//   C/D (16x16 f32): VGPR j -> M = j + 8*(lane>=16), N = lane&15.
__device__ __forceinline__ v16h load_a_frag(const float* __restrict__ xrow,
                                            int half_id) {
  const int ka0 = half_id ? 16 : 0;
  const int ka1 = ka0 + 8;
  const float4 a0 = *(const float4*)(xrow + ka0);
  const float4 a1 = *(const float4*)(xrow + ka0 + 4);
  const float4 a2 = *(const float4*)(xrow + ka1);
  const float4 a3 = *(const float4*)(xrow + ka1 + 4);
  v16h a;
  a[0]=(_Float16)a0.x;  a[1]=(_Float16)a0.y;  a[2]=(_Float16)a0.z;  a[3]=(_Float16)a0.w;
  a[4]=(_Float16)a1.x;  a[5]=(_Float16)a1.y;  a[6]=(_Float16)a1.z;  a[7]=(_Float16)a1.w;
  a[8]=(_Float16)a2.x;  a[9]=(_Float16)a2.y;  a[10]=(_Float16)a2.z; a[11]=(_Float16)a2.w;
  a[12]=(_Float16)a3.x; a[13]=(_Float16)a3.y; a[14]=(_Float16)a3.z; a[15]=(_Float16)a3.w;
  return a;
}

__device__ __forceinline__ v16h load_b_frag(const float* __restrict__ wrow) {
  // wrow points at W[col*K + k0 + (half_id?16:0)]; 16 contiguous floats.
  const float4 b0 = *(const float4*)(wrow);
  const float4 b1 = *(const float4*)(wrow + 4);
  const float4 b2 = *(const float4*)(wrow + 8);
  const float4 b3 = *(const float4*)(wrow + 12);
  v16h b;
  b[0]=(_Float16)b0.x;  b[1]=(_Float16)b0.y;  b[2]=(_Float16)b0.z;  b[3]=(_Float16)b0.w;
  b[4]=(_Float16)b1.x;  b[5]=(_Float16)b1.y;  b[6]=(_Float16)b1.z;  b[7]=(_Float16)b1.w;
  b[8]=(_Float16)b2.x;  b[9]=(_Float16)b2.y;  b[10]=(_Float16)b2.z; b[11]=(_Float16)b2.w;
  b[12]=(_Float16)b3.x; b[13]=(_Float16)b3.y; b[14]=(_Float16)b3.z; b[15]=(_Float16)b3.w;
  return b;
}

// ---------------------------------------------------------------------------
// GEMM1 (fused): XL = X@Wl^T, XR = X@Wr^T.  X:[M,32], Wl/Wr:[256,32] row-major.
// One wave per (16-row, 32-col) strip: A fragment loaded ONCE, then 4 WMMAs
// (2 matrices x 2 column tiles). grid = (M/16, 256/32), block = 32.
__global__ void __launch_bounds__(32, 1)
gemm_xlxr_kernel(const float* __restrict__ X,
                 const float* __restrict__ Wl,
                 const float* __restrict__ Wr,
                 float* __restrict__ XL,
                 float* __restrict__ XR) {
  const int lane    = threadIdx.x;
  const int half_id = lane >> 4;
  const int l       = lane & 15;
  const int tm      = blockIdx.x;
  const int tnBase  = blockIdx.y * 2;     // column tile base (2 tiles/strip)

  const int row = tm * 16 + l;
  const v16h a = load_a_frag(X + (size_t)row * DIN, half_id);

#pragma unroll
  for (int m = 0; m < 2; ++m) {
    const float* __restrict__ W = m ? Wr : Wl;
    float* __restrict__ Y       = m ? XR : XL;
#pragma unroll
    for (int t = 0; t < 2; ++t) {
      const int col = (tnBase + t) * 16 + l;
      const v16h b = load_b_frag(W + (size_t)col * DIN + (half_id ? 16 : 0));
      v8f acc = {};
      acc = __builtin_amdgcn_wmma_f32_16x16x32_f16(
          false, a, false, b, (short)0, acc, false, false);
      float* yp = Y + (size_t)(tm * 16 + 8 * half_id) * DHID + (tnBase + t) * 16 + l;
#pragma unroll
      for (int j = 0; j < 8; ++j) yp[(size_t)j * DHID] = acc[j];
    }
  }
}

// ---------------------------------------------------------------------------
// GEMM2: Y = X@W^T.  X:[M,64], W:[128,64] row-major, Y:[M,128].
// One wave per (16-row, 32-col) strip: per k-step A is loaded once and reused
// across 2 column tiles (2 accumulators, 4 WMMAs total). grid = (M/16, 4).
__global__ void __launch_bounds__(32, 1)
gemm_h_kernel(const float* __restrict__ X,
              const float* __restrict__ W,
              float* __restrict__ Y) {
  const int lane    = threadIdx.x;
  const int half_id = lane >> 4;
  const int l       = lane & 15;
  const int tm      = blockIdx.x;
  const int tnBase  = blockIdx.y * 2;

  const int row = tm * 16 + l;
  v8f acc0 = {};
  v8f acc1 = {};

#pragma unroll
  for (int k0 = 0; k0 < NC; k0 += 32) {
    const v16h a = load_a_frag(X + (size_t)row * NC + k0, half_id);
    const int kb = k0 + (half_id ? 16 : 0);
    const v16h b0 = load_b_frag(W + (size_t)(tnBase * 16 + l) * NC + kb);
    const v16h b1 = load_b_frag(W + (size_t)((tnBase + 1) * 16 + l) * NC + kb);
    acc0 = __builtin_amdgcn_wmma_f32_16x16x32_f16(
        false, a, false, b0, (short)0, acc0, false, false);
    acc1 = __builtin_amdgcn_wmma_f32_16x16x32_f16(
        false, a, false, b1, (short)0, acc1, false, false);
  }
  float* yp0 = Y + (size_t)(tm * 16 + 8 * half_id) * DOUT + tnBase * 16 + l;
  float* yp1 = yp0 + 16;
#pragma unroll
  for (int j = 0; j < 8; ++j) {
    yp0[(size_t)j * DOUT] = acc0[j];
    yp1[(size_t)j * DOUT] = acc1[j];
  }
}

// ---------------------------------------------------------------------------
// One wave per edge. Lane layout: head h = lane>>3, 8 channels per lane.
// logits[e,h] = sum_c att[h,c] * leaky_relu(xl[src,h,c] + xr[dst,h,c])
// Also: atomicMax of encoded logit into m_enc[dst,h]; degree count.
__global__ void edge_logits_kernel(const long long* __restrict__ ei,
                                   const float* __restrict__ xl,
                                   const float* __restrict__ xr,
                                   const float* __restrict__ att,
                                   float* __restrict__ logits,
                                   unsigned* __restrict__ m_enc,
                                   float* __restrict__ deg, int E) {
  const int wid  = blockIdx.x * (blockDim.x >> 5) + (threadIdx.x >> 5);
  const int lane = threadIdx.x & 31;
  if (wid >= E) return;
  const long long src = ei[wid];
  const long long dst = ei[(size_t)E + wid];

  const int h  = lane >> 3;
  const int cg = (lane & 7) * 8;
  const size_t co = (size_t)h * NC + cg;

  const float4 l0 = *(const float4*)(xl + (size_t)src * DHID + co);
  const float4 l1 = *(const float4*)(xl + (size_t)src * DHID + co + 4);
  const float4 r0 = *(const float4*)(xr + (size_t)dst * DHID + co);
  const float4 r1 = *(const float4*)(xr + (size_t)dst * DHID + co + 4);
  const float4 a0 = *(const float4*)(att + co);
  const float4 a1 = *(const float4*)(att + co + 4);

  float vs[8] = {l0.x + r0.x, l0.y + r0.y, l0.z + r0.z, l0.w + r0.w,
                 l1.x + r1.x, l1.y + r1.y, l1.z + r1.z, l1.w + r1.w};
  const float as[8] = {a0.x, a0.y, a0.z, a0.w, a1.x, a1.y, a1.z, a1.w};
  float part = 0.0f;
#pragma unroll
  for (int i = 0; i < 8; ++i) {
    const float v = (vs[i] > 0.0f) ? vs[i] : NEG_SLOPE * vs[i];
    part += as[i] * v;
  }
  // reduce across the 8 lanes of one head
  part += __shfl_xor(part, 1, 32);
  part += __shfl_xor(part, 2, 32);
  part += __shfl_xor(part, 4, 32);

  if ((lane & 7) == 0) {
    logits[(size_t)wid * NH + h] = part;
    atomicMax(&m_enc[(size_t)dst * NH + h], enc_f32(part));
  }
  if (lane == 0) atomicAdd(&deg[dst], 1.0f);
}

// ---------------------------------------------------------------------------
// One thread per (edge, head): e = exp(logit - max); accumulate segment sum.
__global__ void exp_sum_kernel(const long long* __restrict__ ei,
                               float* __restrict__ logits,
                               const unsigned* __restrict__ m_enc,
                               float* __restrict__ s, int E) {
  const long idx = (long)blockIdx.x * blockDim.x + threadIdx.x;
  if (idx >= (long)E * NH) return;
  const int e = (int)(idx >> 2);
  const int h = (int)(idx & 3);
  const long long dst = ei[(size_t)E + e];
  const float m = dec_f32(m_enc[(size_t)dst * NH + h]);
  const float v = __expf(logits[idx] - m);
  logits[idx] = v;  // overwrite logits with exp values
  atomicAdd(&s[(size_t)dst * NH + h], v);
}

// ---------------------------------------------------------------------------
// Per node: dinv = deg > 0 ? rsqrt(deg) : 0
__global__ void node_prep_kernel(const float* __restrict__ deg,
                                 float* __restrict__ dinv, int n) {
  const int i = blockIdx.x * blockDim.x + threadIdx.x;
  if (i >= n) return;
  const float d = deg[i];
  dinv[i] = (d > 0.0f) ? rsqrtf(d) : 0.0f;
}

// ---------------------------------------------------------------------------
// One wave per edge: accum[dst, h, c] += xl[src, h, c] * alpha[e, h]
__global__ void scatter_msg_kernel(const long long* __restrict__ ei,
                                   const float* __restrict__ xl,
                                   const float* __restrict__ evals,
                                   const float* __restrict__ s,
                                   float* __restrict__ accum, int E) {
  const int wid  = blockIdx.x * (blockDim.x >> 5) + (threadIdx.x >> 5);
  const int lane = threadIdx.x & 31;
  if (wid >= E) return;
  const long long src = ei[wid];
  const long long dst = ei[(size_t)E + wid];
  const int h  = lane >> 3;
  const int cg = (lane & 7) * 8;
  const size_t co = (size_t)h * NC + cg;

  const float alpha = evals[(size_t)wid * NH + h] /
                      (s[(size_t)dst * NH + h] + 1e-16f);
  const float4 p0 = *(const float4*)(xl + (size_t)src * DHID + co);
  const float4 p1 = *(const float4*)(xl + (size_t)src * DHID + co + 4);
  float* pd = accum + (size_t)dst * DHID + co;
  const float m[8] = {p0.x * alpha, p0.y * alpha, p0.z * alpha, p0.w * alpha,
                      p1.x * alpha, p1.y * alpha, p1.z * alpha, p1.w * alpha};
#pragma unroll
  for (int i = 0; i < 8; ++i) atomicAdd(&pd[i], m[i]);
}

// ---------------------------------------------------------------------------
// Per (node, channel): head mean then ELU -> h [N, 64]
__global__ void mean_elu_kernel(const float* __restrict__ accum,
                                float* __restrict__ hout, int n) {
  const long idx = (long)blockIdx.x * blockDim.x + threadIdx.x;
  if (idx >= (long)n * NC) return;
  const long nd = idx >> 6;
  const int  c  = (int)(idx & 63);
  const float* p = accum + (size_t)nd * DHID + c;
  float v = 0.25f * (p[0] + p[NC] + p[2 * NC] + p[3 * NC]);
  hout[idx] = (v > 0.0f) ? v : (__expf(v) - 1.0f);
}

// ---------------------------------------------------------------------------
// One wave per edge, 4 channels/lane: out[dst,:] += hw[src,:] * dinv[s]*dinv[d]
__global__ void gcn_scatter_kernel(const long long* __restrict__ ei,
                                   const float* __restrict__ hw,
                                   const float* __restrict__ dinv,
                                   float* __restrict__ out, int E) {
  const int wid  = blockIdx.x * (blockDim.x >> 5) + (threadIdx.x >> 5);
  const int lane = threadIdx.x & 31;
  if (wid >= E) return;
  const long long src = ei[wid];
  const long long dst = ei[(size_t)E + wid];
  const float norm = dinv[src] * dinv[dst];
  const int c0 = lane * 4;
  const float4 ph = *(const float4*)(hw + (size_t)src * DOUT + c0);
  float* po = out + (size_t)dst * DOUT + c0;
  atomicAdd(&po[0], ph.x * norm);
  atomicAdd(&po[1], ph.y * norm);
  atomicAdd(&po[2], ph.z * norm);
  atomicAdd(&po[3], ph.w * norm);
}

// ---------------------------------------------------------------------------
extern "C" void kernel_launch(void* const* d_in, const int* in_sizes, int n_in,
                              void* d_out, int out_size, void* d_ws, size_t ws_size,
                              hipStream_t stream) {
  (void)n_in; (void)out_size; (void)ws_size;

  const float*     x   = (const float*)d_in[0];      // [N, 32]
  const long long* ei  = (const long long*)d_in[1];  // [2, E] int64
  const float*     Wl  = (const float*)d_in[2];      // [256, 32]
  const float*     Wr  = (const float*)d_in[3];      // [256, 32]
  const float*     att = (const float*)d_in[4];      // [4, 64]
  const float*     Wg  = (const float*)d_in[5];      // [128, 64]
  float*           out = (float*)d_out;              // [N, 128]

  const int N = in_sizes[0] / DIN;      // 50000
  const int E = in_sizes[1] / 2;        // 800000

  // ---- workspace layout (floats) ----
  float* ws = (float*)d_ws;
  size_t off = 0;
  float*    xl     = ws + off; off += (size_t)N * DHID;   // 12.8M
  float*    xr     = ws + off; off += (size_t)N * DHID;   // 12.8M
  float*    logits = ws + off; off += (size_t)E * NH;     // 3.2M (logits->exp)
  // zero-initialized contiguous span starts here:
  float*    zspan  = ws + off;
  unsigned* m_enc  = (unsigned*)(ws + off); off += (size_t)N * NH; // 200k
  float*    s      = ws + off; off += (size_t)N * NH;     // 200k
  float*    deg    = ws + off; off += (size_t)N;          // 50k
  float*    accum  = ws + off; off += (size_t)N * DHID;   // 12.8M
  const long zcount = (long)(ws + off - zspan);
  float*    dinv   = ws + off; off += (size_t)N;
  float*    hbuf   = ws + off; off += (size_t)N * NC;     // 3.2M
  float*    hw     = ws + off; off += (size_t)N * DOUT;   // 6.4M

  const int mt = N / 16;                 // 3125 row tiles (50000 % 16 == 0)
  const int wavesPerBlk = 8;             // 256 threads = 8 wave32
  const int edgeBlocks = (E + wavesPerBlk - 1) / wavesPerBlk;

  // 1) zero accumulators (m_enc/s/deg/accum) and the output accumulator
  zero_f32_kernel<<<2048, 256, 0, stream>>>(zspan, zcount);
  zero_f32_kernel<<<2048, 256, 0, stream>>>(out, (long)N * DOUT);

  // 2) xl = x @ Wl^T and xr = x @ Wr^T, fused: A loaded once, 4 WMMAs/wave
  gemm_xlxr_kernel<<<dim3(mt, DHID / 32), 32, 0, stream>>>(x, Wl, Wr, xl, xr);

  // 3) attention logits + segment max + degrees
  edge_logits_kernel<<<edgeBlocks, 256, 0, stream>>>(ei, xl, xr, att, logits,
                                                     m_enc, deg, E);
  // 4) exp(logit - max) + segment sum
  exp_sum_kernel<<<(int)(((long)E * NH + 255) / 256), 256, 0, stream>>>(
      ei, logits, m_enc, s, E);

  // 5) dinv = rsqrt(deg)
  node_prep_kernel<<<(N + 255) / 256, 256, 0, stream>>>(deg, dinv, N);

  // 6) alpha-weighted message scatter
  scatter_msg_kernel<<<edgeBlocks, 256, 0, stream>>>(ei, xl, logits, s, accum, E);

  // 7) head mean + ELU
  mean_elu_kernel<<<(int)(((long)N * NC + 255) / 256), 256, 0, stream>>>(
      accum, hbuf, N);

  // 8) hw = h @ Wg^T (K=64, A reused across 2 column tiles, 4 WMMAs/wave)
  gemm_h_kernel<<<dim3(mt, DOUT / 32), 32, 0, stream>>>(hbuf, Wg, hw);

  // 9) degree-normalized GCN scatter into d_out
  gcn_scatter_kernel<<<edgeBlocks, 256, 0, stream>>>(ei, hw, dinv, out, E);
}
